// _CausalSelfAttention_42356967473319
// MI455X (gfx1250) — matrix operand order repaired and checked by reading, not compiled, not run on hardware
//
#include <hip/hip_runtime.h>
#include <stdint.h>

#define D_MODEL 1024
#define N_HEADS 16
#define HEAD_DIM 64
#define BATCH 4
#define SEQ 2048
#define ROWS (BATCH * SEQ)   // 8192
#define QKV_N (3 * D_MODEL)  // 3072

// LDS panel geometry: 128 rows x 32 bf16, rows padded to 40 elem (80B) for banks
#define BM 128
#define BN 128
#define BK 32
#define LDS_STRIDE 40                       // elements per row (32 data + 8 pad)
#define PANEL (BM * LDS_STRIDE)             // elements per panel
#define SMEM_ELEMS (4 * PANEL)              // A(2 bufs) + B(2 bufs) = 40 KiB

typedef __attribute__((ext_vector_type(16))) __bf16 v16bf_t;
typedef __attribute__((ext_vector_type(8)))  float  v8f_t;

union FragB16 {
  v16bf_t v;
  unsigned int u[8];
  unsigned short h[16];
};

__device__ __forceinline__ unsigned short f32_to_bf16(float f) {
  unsigned int u = __float_as_uint(f);
  u += 0x7FFFu + ((u >> 16) & 1u);  // round-to-nearest-even
  return (unsigned short)(u >> 16);
}

// B-fragment: 16 contiguous bf16 per lane (lane's column fixed, K contiguous)
__device__ __forceinline__ void load_b_frag(FragB16& b, const unsigned short* p) {
  const uint4* bp = (const uint4*)p;
  uint4 lo = bp[0], hi = bp[1];
  b.u[0] = lo.x; b.u[1] = lo.y; b.u[2] = lo.z; b.u[3] = lo.w;
  b.u[4] = hi.x; b.u[5] = hi.y; b.u[6] = hi.z; b.u[7] = hi.w;
}

// A-fragment pairs: h[0..15] = K(16*half)..+15 (contiguous)
__device__ __forceinline__ void load_a_frag(FragB16& a, const unsigned short* p) {
  const unsigned int* ap = (const unsigned int*)p;
#pragma unroll
  for (int j = 0; j < 4; ++j) { a.u[j] = ap[j]; a.u[4 + j] = ap[4 + j]; }
}

// ---------------------------------------------------------------- prep
__global__ void k_cvt_bf16(const float* __restrict__ in,
                           unsigned short* __restrict__ out, int n) {
  int i = blockIdx.x * blockDim.x + threadIdx.x;
  if (i < n) out[i] = f32_to_bf16(in[i]);
}

// Coalesced LDS-tiled transpose: W[k*N+n] (f32) -> Wt[n*K+k] (bf16).
// 256 threads handle a 32x32 tile; reads 128B-coalesced, writes 64B-coalesced.
__global__ __launch_bounds__(256) void k_transpose_bf16(
    const float* __restrict__ W, unsigned short* __restrict__ Wt, int K, int N) {
  __shared__ unsigned short tile[32][33];
  const int n0 = blockIdx.x * 32, k0 = blockIdx.y * 32;
  const int tx = threadIdx.x & 31, ty = threadIdx.x >> 5;  // ty 0..7
#pragma unroll
  for (int s = 0; s < 4; ++s) {
    int kk = ty + 8 * s;
    tile[kk][tx] = f32_to_bf16(W[(size_t)(k0 + kk) * N + n0 + tx]);
  }
  __syncthreads();
#pragma unroll
  for (int s = 0; s < 4; ++s) {
    int r = ty + 8 * s;
    Wt[(size_t)(n0 + r) * K + k0 + tx] = tile[tx][r];
  }
}

// --------------------------------------------- async LDS panel fill
// 256 threads cooperatively DMA a 128x32 bf16 panel (rows padded to 80B) from
// global (row-major, row stride K elements) into LDS via the CDNA5 async path.
// Per wave: 2 instructions -> ASYNCcnt += 2.
__device__ __forceinline__ void async_fill_panel(
    const unsigned short* __restrict__ gbase, int K, int row0, int k0,
    unsigned ldsBase) {
  const int t = threadIdx.x;
#pragma unroll
  for (int i = 0; i < 2; ++i) {
    int seg = t + 256 * i;     // 0..511 : 128 rows x 4 x 16B segments
    int row = seg >> 2;
    int off16 = seg & 3;
    unsigned vofs  = (unsigned)(((row0 + row) * K + k0 + off16 * 8) * 2);
    unsigned laddr = ldsBase + (unsigned)(row * (LDS_STRIDE * 2) + off16 * 16);
    asm volatile("global_load_async_to_lds_b128 %0, %1, %2"
                 :: "v"(laddr), "v"(vofs), "s"(gbase) : "memory");
  }
}

#define WAIT_ASYNC_LE(n) asm volatile("s_wait_asynccnt " #n ::: "memory")

// --------------------------------------------- cooperative GEMM core
// 256-thread block computes a 128x128 tile of A[MxK] * Bt[NxK]^T.
// Double-buffered LDS panels fed by async DMA; 8 waves each own a 32x64
// register tile (8 accumulators = 8 WMMA per K-chunk per wave).
__device__ __forceinline__ void block_gemm_128x128(
    const unsigned short* __restrict__ A, const unsigned short* __restrict__ Bt,
    int K, int m0, int n0, unsigned short* smem, v8f_t acc[2][4]) {
  const int lane = threadIdx.x & 31;
  const int half = lane >> 4;
  const int l16  = lane & 15;
  const int w    = threadIdx.x >> 5;
  const int wm   = (w & 3) * 32;   // wave's row offset inside tile
  const int wn   = (w >> 2) * 64;  // wave's col offset inside tile

  unsigned short* Abuf = smem;
  unsigned short* Bbuf = smem + 2 * PANEL;
  const unsigned aLds = (unsigned)(size_t)(void*)Abuf;  // low 32b = LDS offset
  const unsigned bLds = (unsigned)(size_t)(void*)Bbuf;

  const int NC = K / BK;
  async_fill_panel(A,  K, m0, 0, aLds);   // prologue fill, ASYNCcnt += 4
  async_fill_panel(Bt, K, n0, 0, bLds);

  for (int c = 0; c < NC; ++c) {
    const int p = c & 1;
    if (c + 1 < NC) {  // prefetch next chunk into other buffer
      async_fill_panel(A,  K, m0, (c + 1) * BK, aLds + (p ^ 1) * PANEL * 2);
      async_fill_panel(Bt, K, n0, (c + 1) * BK, bLds + (p ^ 1) * PANEL * 2);
      WAIT_ASYNC_LE(4);  // only the newer 4 fills may remain outstanding
    } else {
      WAIT_ASYNC_LE(0);
    }
    __syncthreads();  // all waves' DMA for buffer p complete

    const unsigned short* At = Abuf + p * PANEL;
    const unsigned short* Bp = Bbuf + p * PANEL;
    FragB16 a[2];
#pragma unroll
    for (int i = 0; i < 2; ++i)
      load_a_frag(a[i], At + (wm + 16 * i + l16) * LDS_STRIDE + 16 * half);
#pragma unroll
    for (int j = 0; j < 4; ++j) {
      FragB16 b;
      load_b_frag(b, Bp + (wn + 16 * j + l16) * LDS_STRIDE + 16 * half);
#pragma unroll
      for (int i = 0; i < 2; ++i)
        acc[i][j] = __builtin_amdgcn_wmma_f32_16x16x32_bf16(
            false, a[i].v, false, b.v, (short)0, acc[i][j], false, false);
    }
    __syncthreads();  // reads done before buffer p is refilled next iter
  }
}

// ------------------------------------------------------------ QKV GEMM
__global__ __launch_bounds__(256) void k_qkv_gemm(
    const unsigned short* __restrict__ xb,   // [8192 x 1024] bf16
    const unsigned short* __restrict__ wt,   // [3072 x 1024] bf16 (N-major)
    unsigned short* __restrict__ qb,         // [B,H,T,Dh]
    unsigned short* __restrict__ kb,         // [B,H,T,Dh]
    unsigned short* __restrict__ vt) {       // [B,H,Dh,T]
  __shared__ __align__(16) unsigned short smem[SMEM_ELEMS];
  const int m0 = blockIdx.x * BM;
  const int n0 = blockIdx.y * BN;
  v8f_t zero = {0.f, 0.f, 0.f, 0.f, 0.f, 0.f, 0.f, 0.f};
  v8f_t acc[2][4] = {{zero, zero, zero, zero}, {zero, zero, zero, zero}};
  block_gemm_128x128(xb, wt, D_MODEL, m0, n0, smem, acc);

  const int lane = threadIdx.x & 31;
  const int half = lane >> 4, l16 = lane & 15;
  const int w  = threadIdx.x >> 5;
  const int wm = (w & 3) * 32, wn = (w >> 2) * 64;
  const int gc0 = n0 + wn;            // 64-wide: single section & head per wave
  const int sec = gc0 >> 10;          // 0:q 1:k 2:v
  const int h   = (gc0 & 1023) >> 6;

  if (sec == 2) {
    // V: stage wave's 32(t) x 64(dh) tile transposed in per-wave LDS slice
    // (panel LDS is dead after block_gemm's final barrier), then store the
    // [dh][t] layout with coalesced b128 writes.
    unsigned short* tbuf = smem + w * 2048;  // 64 dh rows x 32 t = 4 KiB
#pragma unroll
    for (int i = 0; i < 2; ++i)
#pragma unroll
      for (int j = 0; j < 4; ++j)
#pragma unroll
        for (int r = 0; r < 8; ++r) {
          int tl = 16 * i + r + 8 * half;
          int dh = 16 * j + l16;
          tbuf[dh * 32 + tl] = f32_to_bf16(acc[i][j][r]);
        }
    asm volatile("s_wait_dscnt 0" ::: "memory");  // in-wave cross-lane LDS dep
    int b  = (m0 + wm) >> 11;
    int t0 = (m0 + wm) & (SEQ - 1);
    size_t bh = (size_t)(b * N_HEADS + h);
#pragma unroll
    for (int pass = 0; pass < 8; ++pass) {
      int idx = pass * 32 + lane;        // 256 x 16B segments
      int dh = idx >> 2, seg = idx & 3;
      uint4 val = *(const uint4*)(tbuf + dh * 32 + seg * 8);
      *(uint4*)(vt + (bh * HEAD_DIM + dh) * SEQ + t0 + seg * 8) = val;
    }
  } else {
#pragma unroll
    for (int i = 0; i < 2; ++i) {
#pragma unroll
      for (int j = 0; j < 4; ++j) {
#pragma unroll
        for (int r = 0; r < 8; ++r) {
          int grow = m0 + wm + 16 * i + r + 8 * half;
          int b = grow >> 11, t = grow & (SEQ - 1);
          int dh = 16 * j + l16;
          unsigned short bv = f32_to_bf16(acc[i][j][r]);
          size_t bh = (size_t)(b * N_HEADS + h);
          if (sec == 0) qb[(bh * SEQ + t) * HEAD_DIM + dh] = bv;
          else          kb[(bh * SEQ + t) * HEAD_DIM + dh] = bv;
        }
      }
    }
  }
}

// ----------------------------------------------------- output projection
__global__ __launch_bounds__(256) void k_out_gemm(
    const unsigned short* __restrict__ ab,  // att [8192 x 1024] bf16
    const unsigned short* __restrict__ wt,  // W_out^T [1024 x 1024] bf16
    float* __restrict__ y) {                // [8192 x 1024] f32
  __shared__ __align__(16) unsigned short smem[SMEM_ELEMS];
  const int m0 = blockIdx.x * BM;
  const int n0 = blockIdx.y * BN;
  v8f_t zero = {0.f, 0.f, 0.f, 0.f, 0.f, 0.f, 0.f, 0.f};
  v8f_t acc[2][4] = {{zero, zero, zero, zero}, {zero, zero, zero, zero}};
  block_gemm_128x128(ab, wt, D_MODEL, m0, n0, smem, acc);

  const int lane = threadIdx.x & 31;
  const int half = lane >> 4, l16 = lane & 15;
  const int w  = threadIdx.x >> 5;
  const int wm = (w & 3) * 32, wn = (w >> 2) * 64;
#pragma unroll
  for (int i = 0; i < 2; ++i)
#pragma unroll
    for (int j = 0; j < 4; ++j)
#pragma unroll
      for (int r = 0; r < 8; ++r)
        y[(size_t)(m0 + wm + 16 * i + r + 8 * half) * D_MODEL +
          n0 + wn + 16 * j + l16] = acc[i][j][r];
}

// ------------------------------------------------- flash attention core
__global__ __launch_bounds__(128) void k_attn(
    const unsigned short* __restrict__ q,   // [B,H,T,Dh] bf16
    const unsigned short* __restrict__ kk,  // [B,H,T,Dh] bf16
    const unsigned short* __restrict__ vt,  // [B,H,Dh,T] bf16
    unsigned short* __restrict__ att) {     // [B*T, C] bf16
  __shared__ unsigned short ldsP[4 * 16 * 32];  // per-wave 16x32 P staging
  const int wave = threadIdx.x >> 5;
  const int lane = threadIdx.x & 31;
  const int half = lane >> 4;
  const int l16  = lane & 15;
  const int QT_PER_BH = SEQ / 16;  // 128
  int gw = blockIdx.x * 4 + wave;
  int bh = gw / QT_PER_BH;
  int qt = gw - bh * QT_PER_BH;
  int qbase = qt * 16;

  const unsigned short* qp = q  + (size_t)bh * SEQ * HEAD_DIM;
  const unsigned short* kp = kk + (size_t)bh * SEQ * HEAD_DIM;
  const unsigned short* vp = vt + (size_t)bh * HEAD_DIM * SEQ;
  unsigned short* lp = ldsP + wave * (16 * 32);

  FragB16 aq[2];
  {
    const size_t rbase = (size_t)(qbase + l16) * HEAD_DIM;
#pragma unroll
    for (int c = 0; c < 2; ++c)
      load_a_frag(aq[c], qp + rbase + 32 * c + 16 * half);
  }

  v8f_t zero = {0.f, 0.f, 0.f, 0.f, 0.f, 0.f, 0.f, 0.f};
  v8f_t o[4] = {zero, zero, zero, zero};
  float mrow[8], lrow[8];
#pragma unroll
  for (int r = 0; r < 8; ++r) { mrow[r] = -1e30f; lrow[r] = 0.0f; }

  const int qend = qbase + 15;
  for (int kb2 = 0; kb2 <= qend; kb2 += 32) {
    v8f_t S[2];
#pragma unroll
    for (int s = 0; s < 2; ++s) {
      FragB16 b0, b1;
      const size_t krow = (size_t)(kb2 + 16 * s + l16) * HEAD_DIM + 16 * half;
      load_b_frag(b0, kp + krow);
      load_b_frag(b1, kp + krow + 32);
      v8f_t sacc = __builtin_amdgcn_wmma_f32_16x16x32_bf16(
          false, aq[0].v, false, b0.v, (short)0, zero, false, false);
      S[s] = __builtin_amdgcn_wmma_f32_16x16x32_bf16(
          false, aq[1].v, false, b1.v, (short)0, sacc, false, false);
    }

    float cm[8];
#pragma unroll
    for (int r = 0; r < 8; ++r) {
      int row = qbase + r + 8 * half;
      float v0 = S[0][r] * 0.125f;
      float v1 = S[1][r] * 0.125f;
      if (kb2 + l16 > row)      v0 = -1e30f;
      if (kb2 + 16 + l16 > row) v1 = -1e30f;
      S[0][r] = v0; S[1][r] = v1;
      cm[r] = fmaxf(v0, v1);
    }
#pragma unroll
    for (int r = 0; r < 8; ++r)
      for (int msk = 1; msk <= 8; msk <<= 1)
        cm[r] = fmaxf(cm[r], __shfl_xor(cm[r], msk, 32));

#pragma unroll
    for (int r = 0; r < 8; ++r) {
      float mn = fmaxf(mrow[r], cm[r]);
      float sc = __expf(mrow[r] - mn);
      mrow[r] = mn;
      float p0 = __expf(S[0][r] - mn);
      float p1 = __expf(S[1][r] - mn);
      float rs = p0 + p1;
      for (int msk = 1; msk <= 8; msk <<= 1)
        rs += __shfl_xor(rs, msk, 32);
      lrow[r] = lrow[r] * sc + rs;
      o[0][r] *= sc; o[1][r] *= sc; o[2][r] *= sc; o[3][r] *= sc;
      int lr = r + 8 * half;
      lp[lr * 32 + l16]      = f32_to_bf16(p0);
      lp[lr * 32 + 16 + l16] = f32_to_bf16(p1);
    }
    asm volatile("s_wait_dscnt 0" ::: "memory");

    FragB16 apf;
    load_a_frag(apf, lp + (size_t)l16 * 32 + 16 * half);
    asm volatile("" ::: "memory");

#pragma unroll
    for (int d = 0; d < 4; ++d) {
      FragB16 bv;
      load_b_frag(bv, vp + (size_t)(16 * d + l16) * SEQ + kb2 + 16 * half);
      o[d] = __builtin_amdgcn_wmma_f32_16x16x32_bf16(
          false, apf.v, false, bv.v, (short)0, o[d], false, false);
    }
  }

  int b = bh >> 4, h = bh & 15;
#pragma unroll
  for (int r = 0; r < 8; ++r) {
    float inv = 1.0f / lrow[r];
    int grow = b * SEQ + qbase + r + 8 * half;
#pragma unroll
    for (int d = 0; d < 4; ++d) {
      int col = h * HEAD_DIM + 16 * d + l16;
      att[(size_t)grow * D_MODEL + col] = f32_to_bf16(o[d][r] * inv);
    }
  }
}

// ------------------------------------------------------------- launcher
extern "C" void kernel_launch(void* const* d_in, const int* in_sizes, int n_in,
                              void* d_out, int out_size, void* d_ws, size_t ws_size,
                              hipStream_t stream) {
  const float* x    = (const float*)d_in[0];
  const float* Wqkv = (const float*)d_in[1];
  const float* Wout = (const float*)d_in[2];
  float* y = (float*)d_out;

  char* ws = (char*)d_ws;
  unsigned short* xb  = (unsigned short*)(ws + 0);         // 16 MiB
  unsigned short* wqt = (unsigned short*)(ws + 16777216);  //  6 MiB
  unsigned short* wot = (unsigned short*)(ws + 23068672);  //  2 MiB
  unsigned short* qb  = (unsigned short*)(ws + 25165824);  // 16 MiB
  unsigned short* kb  = (unsigned short*)(ws + 41943040);  // 16 MiB
  unsigned short* vt  = (unsigned short*)(ws + 58720256);  // 16 MiB
  unsigned short* att = xb;  // x no longer needed after QKV GEMM

  k_cvt_bf16<<<(ROWS * D_MODEL + 255) / 256, 256, 0, stream>>>(x, xb, ROWS * D_MODEL);
  k_transpose_bf16<<<dim3(QKV_N / 32, D_MODEL / 32), 256, 0, stream>>>(Wqkv, wqt, D_MODEL, QKV_N);
  k_transpose_bf16<<<dim3(D_MODEL / 32, D_MODEL / 32), 256, 0, stream>>>(Wout, wot, D_MODEL, D_MODEL);

  // QKV projection: 128x128 tiles over (8192 x 3072)
  k_qkv_gemm<<<dim3(ROWS / BM, QKV_N / BN), 256, 0, stream>>>(xb, wqt, qb, kb, vt);

  // flash attention: 8192 q-tile waves / 4 per block
  k_attn<<<(BATCH * N_HEADS * (SEQ / 16)) / 4, 128, 0, stream>>>(qb, kb, vt, att);

  // output projection: 128x128 tiles over (8192 x 1024)
  k_out_gemm<<<dim3(ROWS / BM, D_MODEL / BN), 256, 0, stream>>>(att, wot, y);
}
